// DifferentiableForest_77610059039210
// MI455X (gfx1250) — matrix operand order repaired and checked by reading, not compiled, not run on hardware
//
#include <hip/hip_runtime.h>
#include <hip/hip_bf16.h>

// ---------------------------------------------------------------------------
// Differentiable forest, fused as one bf16 WMMA GEMM + tree-local epilogue.
//   B=131072, F=256, C=10, T=10, depth=4 (ND=15, L=16)
// Combined weight matrix per tree: 16 (dec, padded) + 160 (leaf l*10+c) = 176
// Pipeline: double-buffered B slices in LDS, async-to-LDS DMA when available.
// ---------------------------------------------------------------------------

typedef __attribute__((ext_vector_type(16))) __bf16 v16bf;
typedef __attribute__((ext_vector_type(8)))  float  v8f;

#define B_ROWS      131072
#define F_DIM       256
#define C_CLS       10
#define T_TREES     10
#define ND_NODES    15
#define TREE_COLS   176          // 16 dec (15 + 1 pad) + 160 leaf
#define NTILE       11           // 176 / 16
#define KSTEPS      8            // 256 / 32
#define NSLICES     (T_TREES * KSTEPS)   // 80
#define M_PER_BLOCK 128
#define WAVES       8
#define TPB         256

#define WT_ELEMS    (T_TREES * TREE_COLS * F_DIM)   // 450560 bf16
#define BIAS_ELEMS  (T_TREES * TREE_COLS)           // 1760 f32

// LDS layout
#define SB_ROW_STRIDE 80                             // bytes per B row (64 used + pad)
#define SB_BYTES      (TREE_COLS * SB_ROW_STRIDE)    // 14080 per buffer
#define SB_TOTAL      (2 * SB_BYTES)                 // double buffered
#define EPI_ROW_DW    180                            // dwords per epilogue row (176 used)
#define EPI_WAVE_DW   (16 * EPI_ROW_DW)              // 2880
#define SEPI_BYTES    (WAVES * EPI_WAVE_DW * 4)      // 92160
#define SBIAS_BYTES   (BIAS_ELEMS * 4)               // 7040
#define SMEM_BYTES    (SB_TOTAL + SEPI_BYTES + SBIAS_BYTES + 64)  // ~127 KB < 320 KB

union AV  { v16bf v; __bf16 e[16]; };
union BVU { v16bf v; uint4 q[2]; };

// Prefer the CDNA5 async global->LDS DMA path when the toolchain declares it.
#if defined(__has_builtin)
#if __has_builtin(__builtin_amdgcn_global_load_async_to_lds_b128) && \
    __has_builtin(__builtin_amdgcn_s_wait_asynccnt)
#define USE_ASYNC_LDS 1
#endif
#endif
#ifndef USE_ASYNC_LDS
#define USE_ASYNC_LDS 0
#endif

// ---------------------------------------------------------------------------
// Prep: pack Wd/Wl -> bf16 Wt[t][176][256], biases -> bias[t][176],
//       softmax(tree_w) -> wsoft[10]
// ---------------------------------------------------------------------------
__global__ void forest_prep_77610059(const float* __restrict__ Wd,
                                     const float* __restrict__ bd,
                                     const float* __restrict__ Wl,
                                     const float* __restrict__ bl,
                                     const float* __restrict__ tw,
                                     __bf16* __restrict__ Wt,
                                     float* __restrict__ bias,
                                     float* __restrict__ wsoft) {
  const int gid = blockIdx.x * blockDim.x + threadIdx.x;
  if (gid < WT_ELEMS) {
    const int t   = gid / (TREE_COLS * F_DIM);
    const int rem = gid - t * TREE_COLS * F_DIM;
    const int row = rem / F_DIM;
    const int k   = rem - row * F_DIM;
    float v;
    if (row < 16) v = (row < ND_NODES) ? Wd[(t * ND_NODES + row) * F_DIM + k] : 0.0f;
    else          v = Wl[(t * 160 + (row - 16)) * F_DIM + k];
    Wt[gid] = (__bf16)v;
  }
  if (gid < BIAS_ELEMS) {
    const int t   = gid / TREE_COLS;
    const int row = gid - t * TREE_COLS;
    float v;
    if (row < 16) v = (row < ND_NODES) ? bd[t * ND_NODES + row] : 0.0f;
    else          v = bl[t * 160 + (row - 16)];
    bias[gid] = v;
  }
  if (gid == 0) {
    float mx = tw[0];
    for (int t = 1; t < T_TREES; ++t) mx = fmaxf(mx, tw[t]);
    float e[T_TREES], s = 0.0f;
    for (int t = 0; t < T_TREES; ++t) { e[t] = __expf(tw[t] - mx); s += e[t]; }
    for (int t = 0; t < T_TREES; ++t) wsoft[t] = e[t] / s;
  }
}

// ---------------------------------------------------------------------------
// B-slice staging helpers. Slice i = (tree i/8, kstep i%8): 176 rows x 64 B.
// Per thread: part = tid&3 (16 B), rows tid>>2, +64, +128 (third if tid<192).
// ---------------------------------------------------------------------------
__device__ __forceinline__ const __bf16* slice_base(const __bf16* Wt, int i) {
  return Wt + (size_t)(i >> 3) * (TREE_COLS * F_DIM) + (i & 7) * 32;
}

#if USE_ASYNC_LDS
// Builtin signature (from hipcc diagnostic): param 1 is a global
// (addrspace(1)) pointer to a 16-byte int vector; param 2 is the LDS-side
// (addrspace(3)) pointer; then (i32 offset, i32 cpol).
typedef int v4i_vec __attribute__((vector_size(16)));
typedef __attribute__((address_space(1))) v4i_vec* gas_ptr;
typedef __attribute__((address_space(3))) v4i_vec* las_ptr;

__device__ __forceinline__ void slice_fetch_async(const __bf16* base, char* buf,
                                                  int tid) {
  const int r = tid >> 2, part = tid & 3;
  __builtin_amdgcn_global_load_async_to_lds_b128(
      (gas_ptr)((char*)(base + r * F_DIM) + part * 16),
      (las_ptr)(buf + r * SB_ROW_STRIDE + part * 16), 0, 0);
  __builtin_amdgcn_global_load_async_to_lds_b128(
      (gas_ptr)((char*)(base + (r + 64) * F_DIM) + part * 16),
      (las_ptr)(buf + (r + 64) * SB_ROW_STRIDE + part * 16), 0, 0);
  if (tid < 192)
    __builtin_amdgcn_global_load_async_to_lds_b128(
        (gas_ptr)((char*)(base + (r + 128) * F_DIM) + part * 16),
        (las_ptr)(buf + (r + 128) * SB_ROW_STRIDE + part * 16), 0, 0);
}
#else
__device__ __forceinline__ void slice_fetch(const __bf16* base, int tid,
                                            uint4& v0, uint4& v1, uint4& v2) {
  const int r = tid >> 2, part = tid & 3;
  v0 = ((const uint4*)(base + r * F_DIM))[part];
  v1 = ((const uint4*)(base + (r + 64) * F_DIM))[part];
  if (tid < 192) v2 = ((const uint4*)(base + (r + 128) * F_DIM))[part];
}
__device__ __forceinline__ void slice_commit(char* buf, int tid,
                                             const uint4& v0, const uint4& v1,
                                             const uint4& v2) {
  const int r = tid >> 2, part = tid & 3;
  *(uint4*)(buf + r * SB_ROW_STRIDE + part * 16) = v0;
  *(uint4*)(buf + (r + 64) * SB_ROW_STRIDE + part * 16) = v1;
  if (tid < 192) *(uint4*)(buf + (r + 128) * SB_ROW_STRIDE + part * 16) = v2;
}
#endif

// ---------------------------------------------------------------------------
// Main fused kernel: 8 waves x 16-row M-tiles, per-tree 11 WMMA N-tiles
// ---------------------------------------------------------------------------
__global__ __launch_bounds__(TPB, 1)
void forest_main_77610059(const float* __restrict__ x,
                          const __bf16* __restrict__ Wt,
                          const float* __restrict__ bias,
                          const float* __restrict__ wsoft,
                          float* __restrict__ out) {
  extern __shared__ char smem[];
  char*  sB    = smem;                                   // 2 x 14080 B
  float* sEpi  = (float*)(smem + SB_TOTAL);
  float* sBias = (float*)(smem + SB_TOTAL + SEPI_BYTES);
  float* sW    = (float*)(smem + SB_TOTAL + SEPI_BYTES + SBIAS_BYTES);

  const int tid  = threadIdx.x;
  const int lane = tid & 31;
  const int wv   = tid >> 5;
  const int half = lane >> 4;   // 0: lanes 0-15, 1: lanes 16-31
  const int l15  = lane & 15;

  // Stage biases + softmaxed tree weights once per block.
  for (int i = tid; i < BIAS_ELEMS; i += TPB) sBias[i] = bias[i];
  if (tid < T_TREES) sW[tid] = wsoft[tid];

  // --- A tile: wave's 16 rows of x, f32 -> bf16, WMMA 16x32 A layout, in regs.
  // Lanes 0-15: row=lane, K {0..7,16..23}+kb ; lanes 16-31: row=lane-16, K {8..15,24..31}+kb
  const int row = blockIdx.x * M_PER_BLOCK + wv * 16 + l15;
  const float* xr = x + (size_t)row * F_DIM;
  const int klo = half * 8;
  AV a[KSTEPS];
#pragma unroll
  for (int ks = 0; ks < KSTEPS; ++ks) {
    const int kb = ks * 32;
    const float4 f0 = *(const float4*)(xr + kb + klo + 0);
    const float4 f1 = *(const float4*)(xr + kb + klo + 4);
    const float4 f2 = *(const float4*)(xr + kb + 16 + klo + 0);
    const float4 f3 = *(const float4*)(xr + kb + 16 + klo + 4);
    a[ks].e[0]  = (__bf16)f0.x; a[ks].e[1]  = (__bf16)f0.y;
    a[ks].e[2]  = (__bf16)f0.z; a[ks].e[3]  = (__bf16)f0.w;
    a[ks].e[4]  = (__bf16)f1.x; a[ks].e[5]  = (__bf16)f1.y;
    a[ks].e[6]  = (__bf16)f1.z; a[ks].e[7]  = (__bf16)f1.w;
    a[ks].e[8]  = (__bf16)f2.x; a[ks].e[9]  = (__bf16)f2.y;
    a[ks].e[10] = (__bf16)f2.z; a[ks].e[11] = (__bf16)f2.w;
    a[ks].e[12] = (__bf16)f3.x; a[ks].e[13] = (__bf16)f3.y;
    a[ks].e[14] = (__bf16)f3.z; a[ks].e[15] = (__bf16)f3.w;
  }

  // Prologue of the B pipeline: stage slice 0 into buffer 0.
#if USE_ASYNC_LDS
  slice_fetch_async(slice_base(Wt, 0), sB, tid);
  __builtin_amdgcn_s_wait_asynccnt(0);
#else
  {
    uint4 v0, v1, v2;
    slice_fetch(slice_base(Wt, 0), tid, v0, v1, v2);
    slice_commit(sB, tid, v0, v1, v2);
  }
#endif
  __syncthreads();

  float outacc[C_CLS];
#pragma unroll
  for (int c = 0; c < C_CLS; ++c) outacc[c] = 0.0f;

  for (int t = 0; t < T_TREES; ++t) {
    const v8f vzero = {0.f, 0.f, 0.f, 0.f, 0.f, 0.f, 0.f, 0.f};
    v8f acc[NTILE];
#pragma unroll
    for (int j = 0; j < NTILE; ++j) acc[j] = vzero;

    for (int ks = 0; ks < KSTEPS; ++ks) {
      const int i = t * KSTEPS + ks;
      char* bufCur = sB + (i & 1) * SB_BYTES;
      char* bufNxt = sB + ((i + 1) & 1) * SB_BYTES;
      const bool have_next = (i + 1) < NSLICES;

      // Issue next slice's global fetches before the math (overlap latency).
#if USE_ASYNC_LDS
      if (have_next) slice_fetch_async(slice_base(Wt, i + 1), bufNxt, tid);
#else
      uint4 v0, v1, v2;
      if (have_next) slice_fetch(slice_base(Wt, i + 1), tid, v0, v1, v2);
#endif

      // 11 WMMAs on current slice; B tile j+1's LDS loads in flight under WMMA j.
      BVU bcur;
      {
        const char* p = bufCur + l15 * SB_ROW_STRIDE + half * 16;
        bcur.q[0] = *(const uint4*)(p);
        bcur.q[1] = *(const uint4*)(p + 32);
      }
#pragma unroll
      for (int j = 0; j < NTILE; ++j) {
        BVU bnxt;
        if (j + 1 < NTILE) {
          const char* p = bufCur + ((j + 1) * 16 + l15) * SB_ROW_STRIDE + half * 16;
          bnxt.q[0] = *(const uint4*)(p);
          bnxt.q[1] = *(const uint4*)(p + 32);
        }
        acc[j] = __builtin_amdgcn_wmma_f32_16x16x32_bf16(
            false, a[ks].v, false, bcur.v, (short)0, acc[j], false, false);
        if (j + 1 < NTILE) bcur = bnxt;
      }

      // Commit next slice into its buffer, then block-sync.
#if USE_ASYNC_LDS
      if (have_next) __builtin_amdgcn_s_wait_asynccnt(0);
#else
      if (have_next) slice_commit(bufNxt, tid, v0, v1, v2);
#endif
      __syncthreads();
    }

    // Dump C tiles (lane=col, VGPR=row) into row-major per-wave LDS region.
    float* epiW = sEpi + wv * EPI_WAVE_DW;
    const int m0 = half * 8;
#pragma unroll
    for (int j = 0; j < NTILE; ++j) {
#pragma unroll
      for (int r = 0; r < 8; ++r)
        epiW[(m0 + r) * EPI_ROW_DW + j * 16 + l15] = acc[j][r];
    }
    __syncthreads();

    // ---- Tree-local epilogue: lane l15 owns row l15; halves split leaves.
    const float* myrow = epiW + l15 * EPI_ROW_DW;
    const float* bT = sBias + t * TREE_COLS;
    const float wt = sW[t];

    float d[ND_NODES];
#pragma unroll
    for (int i2 = 0; i2 < ND_NODES; ++i2) {
      const float z = myrow[i2] + bT[i2];
      d[i2] = 1.0f / (1.0f + __expf(-z));
    }
    // Path recursion exactly as reference: new[k] = old[k%len] *
    //   (k even ? d[s + k/2] : 1 - d[s + k/2])
    float p2[4], p3[8], p4[16];
    {
      const float p1a = d[0], p1b = 1.0f - d[0];
      p2[0] = p1a * d[1]; p2[1] = p1b * (1.0f - d[1]);
      p2[2] = p1a * d[2]; p2[3] = p1b * (1.0f - d[2]);
#pragma unroll
      for (int k = 0; k < 8; ++k)
        p3[k] = p2[k & 3] * ((k & 1) ? (1.0f - d[3 + (k >> 1)]) : d[3 + (k >> 1)]);
#pragma unroll
      for (int k = 0; k < 16; ++k)
        p4[k] = p3[k & 7] * ((k & 1) ? (1.0f - d[7 + (k >> 1)]) : d[7 + (k >> 1)]);
    }
#pragma unroll
    for (int l2 = 0; l2 < 8; ++l2) {
      const float pl = half ? p4[8 + l2] : p4[l2];
      const int l = half * 8 + l2;
      const float* lr = myrow + 16 + l * C_CLS;
      const float* lb = bT + 16 + l * C_CLS;
      float lg[C_CLS];
      float mx = -3.4e38f;
#pragma unroll
      for (int c = 0; c < C_CLS; ++c) { lg[c] = lr[c] + lb[c]; mx = fmaxf(mx, lg[c]); }
      float s = 0.0f;
#pragma unroll
      for (int c = 0; c < C_CLS; ++c) { const float e = __expf(lg[c] - mx); lg[c] = e; s += e; }
      const float scale = pl * wt / s;
#pragma unroll
      for (int c = 0; c < C_CLS; ++c) outacc[c] += lg[c] * scale;
    }
  }

  // Combine the two leaf-halves across the wave (wave32), then store.
#pragma unroll
  for (int c = 0; c < C_CLS; ++c) outacc[c] += __shfl_xor(outacc[c], 16, 32);
  if (half == 0 && row < B_ROWS) {
    float* o = out + (size_t)row * C_CLS;
#pragma unroll
    for (int c = 0; c < C_CLS; ++c) o[c] = outacc[c];
  }
}

// ---------------------------------------------------------------------------
extern "C" void kernel_launch(void* const* d_in, const int* in_sizes, int n_in,
                              void* d_out, int out_size, void* d_ws, size_t ws_size,
                              hipStream_t stream) {
  (void)in_sizes; (void)n_in; (void)out_size; (void)ws_size;
  const float* x  = (const float*)d_in[0];
  const float* Wd = (const float*)d_in[1];
  const float* bd = (const float*)d_in[2];
  const float* Wl = (const float*)d_in[3];
  const float* bl = (const float*)d_in[4];
  const float* tw = (const float*)d_in[5];
  float* out = (float*)d_out;

  // Workspace: Wt bf16 (901120 B) | bias f32 (7040 B) | wsoft f32 (40 B)
  __bf16* Wt    = (__bf16*)d_ws;
  float*  bias  = (float*)((char*)d_ws + (size_t)WT_ELEMS * 2);
  float*  wsoft = bias + BIAS_ELEMS;

  forest_prep_77610059<<<(WT_ELEMS + TPB - 1) / TPB, TPB, 0, stream>>>(
      Wd, bd, Wl, bl, tw, Wt, bias, wsoft);
  forest_main_77610059<<<B_ROWS / M_PER_BLOCK, TPB, SMEM_BYTES, stream>>>(
      x, Wt, bias, wsoft, out);
}
